// HybridCNNGRU_39728447488007
// MI455X (gfx1250) — compile-verified, matrix operand order checked by reading, not run on hardware
//
#include <hip/hip_runtime.h>
#include <hip/hip_bf16.h>
#include <cmath>

// ---------------------------------------------------------------------------
// MI455X (gfx1250): all heavy ops via v_wmma_f32_16x16x32_f16 (wave32 WMMA),
// f16 activations end-to-end, BN folded into conv weights. GEMM A-tiles are
// fetched by the Tensor Data Mover (tensor_load_to_lds, 6-arg toolchain form)
// with the LDS pad feature reproducing the 80B padded row stride, double-
// buffered and synced with s_wait_tensorcnt for copy/compute overlap.
// ---------------------------------------------------------------------------

typedef __attribute__((ext_vector_type(16))) _Float16 v16h;
typedef __attribute__((ext_vector_type(8)))  float    v8f;
typedef __attribute__((ext_vector_type(4)))  unsigned int v4u;
typedef __attribute__((ext_vector_type(8)))  int      v8i;
typedef __attribute__((ext_vector_type(4)))  int      v4i;

union FragH { v16h v; float4 f4[2]; };

__device__ __forceinline__ v8f wmma_f16f32(v16h a, v16h b, v8f c) {
  // 8 args: (neg_a, A, neg_b, B, c_mod, C, reuse_a, reuse_b)
  return __builtin_amdgcn_wmma_f32_16x16x32_f16(false, a, false, b, (short)0, c,
                                                false, false);
}

__device__ __forceinline__ float gelu_exact(float x) {
  return 0.5f * x * (1.0f + erff(x * 0.70710678118654752f));
}

// ---------------------------------------------------------------------------
// TDM: load a [rows<=128] x [32 f16] tile (row stride lda elements) into LDS
// with 16B padding after each 64B row -> LDS row stride 80B (= PAD_LD halves).
// Zero-fills rows/cols beyond (m_rem, k_rem) via tensor-dim OOB handling.
// ---------------------------------------------------------------------------
__device__ __forceinline__ void tdm_load_a_tile(unsigned lds_off,
                                                const _Float16* gptr,
                                                int k_rem, int m_rem, int lda) {
  unsigned long long ga = (unsigned long long)gptr;
  v4u g0 = {};
  g0[0] = 1u;                                   // count=1, user descriptor
  g0[1] = lds_off;                              // lds_addr (bytes)
  g0[2] = (unsigned)(ga & 0xFFFFFFFFu);         // global_addr[31:0]
  g0[3] = (unsigned)((ga >> 32) & 0x01FFFFFFu)  // global_addr[56:32]
        | 0x80000000u;                          // type=2 ("image")
  v8i g1 = {};
  g1[0] = (int)((1u << 16)      // data_size = 2 bytes
              | (1u << 20)      // pad_enable
              | (3u << 22)      // pad_interval: 16 DWORDs (one 64B row)
              | (3u << 25));    // pad_amount:   4 DWORDs (16B)
  unsigned td0 = (unsigned)k_rem;               // tensor_dim0 (remaining K)
  unsigned td1 = (unsigned)m_rem;               // tensor_dim1 (remaining rows)
  g1[1] = (int)((td0 & 0xFFFFu) << 16);                         // [63:48]
  g1[2] = (int)(((td0 >> 16) & 0xFFFFu) | ((td1 & 0xFFFFu) << 16));
  g1[3] = (int)(((td1 >> 16) & 0xFFFFu) | (32u << 16));         // tile_dim0=32
  g1[4] = (int)(128u);                                          // tile_dim1=128
  g1[5] = (int)(unsigned)lda;                   // tensor_dim0_stride[31:0]
  g1[6] = 0;
  g1[7] = 0;
  v4i z4 = {};
  v8i z8 = {};
  // 6-arg toolchain form: (g0 v4u, g1 v8i, g2 v4i, g3 v4i, v8i, cpol)
  __builtin_amdgcn_tensor_load_to_lds(g0, g1, z4, z4, z8, 0);
}

// ---------------------------------------------------------------------------
// Prep kernels
// ---------------------------------------------------------------------------

__global__ void k_build_dct(_Float16* __restrict__ D, _Float16* __restrict__ DT) {
  int idx = blockIdx.x * 256 + threadIdx.x;
  if (idx >= 224 * 224) return;
  int j = idx % 224, i = idx / 224;
  float v = cosf(3.14159265358979323846f * (2.0f * (float)j + 1.0f) * (float)i / 448.0f);
  v *= (i == 0) ? 0.06681531047810609f : 0.09449111825230680f; // sqrt(1/224), sqrt(2/224)
  D[i * 224 + j]  = (_Float16)v;   // D  row-major [h][j]
  DT[j * 224 + i] = (_Float16)v;   // DT row-major [k][n] = D[n][k]
}

__global__ void k_f32_to_f16(const float* __restrict__ in, _Float16* __restrict__ out,
                             long long n) {
  long long i = (long long)blockIdx.x * 256 + threadIdx.x;
  if (i < n) out[i] = (_Float16)in[i];
}

// Fold BatchNorm (eval) into conv weights + bias.
__global__ void k_fold_conv(const float* __restrict__ w, const float* __restrict__ b,
                            const float* __restrict__ g, const float* __restrict__ be,
                            const float* __restrict__ m, const float* __restrict__ v,
                            _Float16* __restrict__ wf, float* __restrict__ bf,
                            int Cout, int Cin) {
  int K = Cin * 9;
  int idx = blockIdx.x * 256 + threadIdx.x;
  if (idx < Cout) {
    float s = g[idx] * rsqrtf(v[idx] + 1e-5f);
    bf[idx] = (b[idx] - m[idx]) * s + be[idx];
  }
  if (idx < Cout * K) {
    int co = idx / K;
    float s = g[co] * rsqrtf(v[co] + 1e-5f);
    wf[idx] = (_Float16)(w[idx] * s);
  }
}

// ---------------------------------------------------------------------------
// Generic batched WMMA GEMM: C = A[M,K] * B[K,N] (+bias)
//   A f16 row-major (TDM-staged).  B f16: btrans==0 -> [K][N]; btrans==1 -> [N][K].
//   mode 0: f16 output; mode 2: f32 output with bias[n].
//   Requires K % 32 == 0 and 16B-aligned rows (all call sites comply).
// ---------------------------------------------------------------------------

#define GT_M 128
#define GT_N 64
#define GT_K 32
#define PAD_LD 40   // f16 elements; 80B LDS row stride (TDM pad reproduces this)

__global__ void __launch_bounds__(256) k_gemm_wmma(
    const _Float16* __restrict__ A, const _Float16* __restrict__ B,
    void* __restrict__ Cp, const float* __restrict__ bias,
    int M, int N, int K, int lda, int ldb, int ldc,
    long long sA_, long long sB_, long long sC_, int mode, int btrans)
{
  __shared__ _Float16 shA[2 * GT_M * PAD_LD];   // double-buffered TDM target
  __shared__ _Float16 shB[GT_N * PAD_LD];
  const int tid = threadIdx.x;
  const int lane = tid & 31, wave = tid >> 5;
  const int l = lane & 15, hi = lane >> 4;
  const int bz = blockIdx.z;
  const _Float16* Ab = A + (long long)bz * sA_;
  const _Float16* Bb = B + (long long)bz * sB_;
  const int m0 = blockIdx.x * GT_M;
  const int n0 = blockIdx.y * GT_N;
  const int mw = (wave & 3) * 32;   // 4 waves along M
  const int nw = (wave >> 2) * 32;  // 2 waves along N

  const unsigned shA_base = (unsigned)(unsigned long long)(void*)&shA[0];
  const _Float16* Atile = Ab + (long long)m0 * lda;
  const int ksteps = K / GT_K;

  v8f acc[2][2] = {};

  // Prologue: TDM fetch of first A tile (wave 0 issues; TDM ignores EXEC).
  if (wave == 0)
    tdm_load_a_tile(shA_base, Atile, K, M - m0, lda);

  for (int kt = 0; kt < ksteps; ++kt) {
    const int k0 = kt * GT_K;
    const _Float16* bufA = shA + (kt & 1) * (GT_M * PAD_LD);

    if (wave == 0) {
      if (kt + 1 < ksteps) {
        tdm_load_a_tile(shA_base + (unsigned)(((kt + 1) & 1) * (GT_M * PAD_LD) * 2),
                        Atile + (k0 + GT_K), K - (k0 + GT_K), M - m0, lda);
        __builtin_amdgcn_s_wait_tensorcnt((short)1);  // current tile landed
      } else {
        __builtin_amdgcn_s_wait_tensorcnt((short)0);
      }
    }

    // Stage B tile (all 256 threads)
    if (btrans) {  // B already [N][K]: direct vector copy
      for (int i = tid; i < (GT_N * GT_K / 8); i += 256) {
        int r = i >> 2, c8 = (i & 3) << 3;
        int gn = n0 + r;
        float4 val = make_float4(0.f, 0.f, 0.f, 0.f);
        if (gn < N) {
          const _Float16* gp = Bb + (long long)gn * ldb + (k0 + c8);
          val = *(const float4*)gp;
          if (k0 + GT_K < K) __builtin_prefetch(gp + GT_K, 0, 3); // global_prefetch_b8
        }
        *(float4*)(shB + r * PAD_LD + c8) = val;
      }
    } else {       // B [K][N]: coalesced read, transposed LDS write
      for (int i = tid; i < (GT_N * GT_K); i += 256) {
        int kk = i >> 6, nn = i & 63;
        int gn = n0 + nn;
        _Float16 vv = (_Float16)0.f;
        if (gn < N) vv = Bb[(long long)(k0 + kk) * ldb + gn];
        shB[nn * PAD_LD + kk] = vv;
      }
    }
    __syncthreads();   // publish TDM A tile + staged B tile to all waves

    FragH fa[2], fb[2];
#pragma unroll
    for (int mi = 0; mi < 2; ++mi) {
      const _Float16* p = bufA + (mw + mi * 16 + l) * PAD_LD + hi * 8;
      fa[mi].f4[0] = *(const float4*)p;          // K chunk {0..7} | {8..15}
      fa[mi].f4[1] = *(const float4*)(p + 16);   // K chunk {16..23} | {24..31}
    }
#pragma unroll
    for (int ni = 0; ni < 2; ++ni) {
      const _Float16* p = shB + (nw + ni * 16 + l) * PAD_LD + hi * 8;
      fb[ni].f4[0] = *(const float4*)p;
      fb[ni].f4[1] = *(const float4*)(p + 16);
    }
#pragma unroll
    for (int mi = 0; mi < 2; ++mi)
#pragma unroll
      for (int ni = 0; ni < 2; ++ni)
        acc[mi][ni] = wmma_f16f32(fa[mi].v, fb[ni].v, acc[mi][ni]);

    __syncthreads();   // everyone done reading before buffers are reused
  }

  // Epilogue: C layout element (vgpr r, lane) -> M = r + (lane>=16?8:0), N = lane&15
#pragma unroll
  for (int mi = 0; mi < 2; ++mi) {
#pragma unroll
    for (int ni = 0; ni < 2; ++ni) {
#pragma unroll
      for (int r = 0; r < 8; ++r) {
        int gm = m0 + mw + mi * 16 + r + hi * 8;
        int gn = n0 + nw + ni * 16 + l;
        if (gm < M && gn < N) {
          float vv = acc[mi][ni][r];
          if (mode == 0) {
            ((_Float16*)Cp)[(long long)bz * sC_ + (long long)gm * ldc + gn] = (_Float16)vv;
          } else {
            if (bias) vv += bias[gn];
            ((float*)Cp)[(long long)bz * sC_ + (long long)gm * ldc + gn] = vv;
          }
        }
      }
    }
  }
}

// ---------------------------------------------------------------------------
// Implicit-GEMM 3x3 SAME conv with folded BN + exact GELU epilogue.
//   M = Cout, N = H*W (per image), K = Cin*9. One image per blockIdx.z.
//   (Halo-gather B staging: not a TDM-friendly access pattern; keep ds path.)
// ---------------------------------------------------------------------------

#define CV_M 64
#define CV_N 128

__global__ void __launch_bounds__(256) k_conv_wmma(
    const _Float16* __restrict__ act, const _Float16* __restrict__ wgt,
    const float* __restrict__ bias, _Float16* __restrict__ out,
    int Cin, int Cout, int H, int W)
{
  __shared__ _Float16 shA[CV_M * PAD_LD];
  __shared__ _Float16 shB[CV_N * PAD_LD];
  const int tid = threadIdx.x;
  const int lane = tid & 31, wave = tid >> 5;
  const int l = lane & 15, hi = lane >> 4;
  const int HW = H * W;
  const int K = Cin * 9;
  const int b = blockIdx.z;
  const int m0 = blockIdx.x * CV_M;
  const int n0 = blockIdx.y * CV_N;
  const int mw = (wave & 1) * 32;   // 2 waves along M
  const int nw = (wave >> 1) * 32;  // 4 waves along N
  const _Float16* actb = act + (long long)b * Cin * HW;

  v8f acc[2][2] = {};
  const int ksteps = (K + GT_K - 1) / GT_K;

  for (int kt = 0; kt < ksteps; ++kt) {
    int k0 = kt * GT_K;
    __syncthreads();
    // Weights tile [64][32]
    for (int i = tid; i < CV_M * GT_K; i += 256) {
      int r = i >> 5, kk = i & 31;
      int gm = m0 + r, gk = k0 + kk;
      _Float16 vv = (_Float16)0.f;
      if (gm < Cout && gk < K) vv = wgt[(long long)gm * K + gk];
      shA[r * PAD_LD + kk] = vv;
    }
    // Activation patch tile [128 spatial][32 k] with SAME-padding gather
    for (int i = tid; i < CV_N * GT_K; i += 256) {
      int nn = i >> 5, kk = i & 31;
      int gk = k0 + kk;
      int p = n0 + nn;
      _Float16 vv = (_Float16)0.f;
      if (p < HW && gk < K) {
        int ci = gk / 9, rs = gk - ci * 9;
        int r3 = rs / 3;
        int dy = r3 - 1, dx = rs - r3 * 3 - 1;
        int py = p / W, px = p - py * W;
        int y = py + dy, x = px + dx;
        if (y >= 0 && y < H && x >= 0 && x < W)
          vv = actb[((long long)ci * H + y) * W + x];
      }
      shB[nn * PAD_LD + kk] = vv;
    }
    __syncthreads();

    FragH fa[2], fb[2];
#pragma unroll
    for (int mi = 0; mi < 2; ++mi) {
      const _Float16* pp = shA + (mw + mi * 16 + l) * PAD_LD + hi * 8;
      fa[mi].f4[0] = *(const float4*)pp;
      fa[mi].f4[1] = *(const float4*)(pp + 16);
    }
#pragma unroll
    for (int ni = 0; ni < 2; ++ni) {
      const _Float16* pp = shB + (nw + ni * 16 + l) * PAD_LD + hi * 8;
      fb[ni].f4[0] = *(const float4*)pp;
      fb[ni].f4[1] = *(const float4*)(pp + 16);
    }
#pragma unroll
    for (int mi = 0; mi < 2; ++mi)
#pragma unroll
      for (int ni = 0; ni < 2; ++ni)
        acc[mi][ni] = wmma_f16f32(fa[mi].v, fb[ni].v, acc[mi][ni]);
  }

#pragma unroll
  for (int mi = 0; mi < 2; ++mi) {
#pragma unroll
    for (int ni = 0; ni < 2; ++ni) {
#pragma unroll
      for (int r = 0; r < 8; ++r) {
        int gm = m0 + mw + mi * 16 + r + hi * 8;
        int gn = n0 + nw + ni * 16 + l;
        if (gm < Cout && gn < HW) {
          float vv = acc[mi][ni][r] + bias[gm];  // BN folded into bias
          vv = gelu_exact(vv);
          out[((long long)b * Cout + gm) * HW + gn] = (_Float16)vv;
        }
      }
    }
  }
}

// ---------------------------------------------------------------------------
// 2x2 average pool, f16 -> f16
// ---------------------------------------------------------------------------
__global__ void k_pool2x2(const _Float16* __restrict__ in, _Float16* __restrict__ out,
                          int H, int W, long long total) {
  long long idx = (long long)blockIdx.x * 256 + threadIdx.x;
  if (idx >= total) return;
  int Wo = W >> 1, Ho = H >> 1;
  int x = (int)(idx % Wo);
  long long t = idx / Wo;
  int y = (int)(t % Ho);
  long long bc = t / Ho;
  const _Float16* p = in + (bc * H + 2 * y) * W + 2 * x;
  float s = (float)p[0] + (float)p[1] + (float)p[W] + (float)p[W + 1];
  out[idx] = (_Float16)(0.25f * s);
}

// [B,512,7,7] -> [B*49, 512]  (transpose(0,2,3,1))
__global__ void k_to_seq(const _Float16* __restrict__ act, _Float16* __restrict__ seq) {
  int idx = blockIdx.x * 256 + threadIdx.x;
  if (idx >= 32 * 49 * 512) return;
  int c = idx & 511;
  int t = idx >> 9;
  int hw = t % 49, bb = t / 49;
  seq[idx] = act[((long long)bb * 512 + c) * 49 + hw];
}

// LayerNorm over last dim; one wave per row; f32 in -> f16 out.
__global__ void __launch_bounds__(256) k_layernorm(
    const float* __restrict__ x, const float* __restrict__ g, const float* __restrict__ b,
    _Float16* __restrict__ out, int rows, int D)
{
  int wave = threadIdx.x >> 5, lane = threadIdx.x & 31;
  int row = blockIdx.x * 8 + wave;
  if (row >= rows) return;
  const float* xr = x + (long long)row * D;
  float s = 0.f, s2 = 0.f;
  for (int j = lane; j < D; j += 32) { float v = xr[j]; s += v; s2 += v * v; }
#pragma unroll
  for (int o = 16; o > 0; o >>= 1) {
    s  += __shfl_xor(s,  o, 32);
    s2 += __shfl_xor(s2, o, 32);
  }
  float mu = s / (float)D;
  float rstd = rsqrtf(fmaxf(s2 / (float)D - mu * mu, 0.f) + 1e-5f);
  for (int j = lane; j < D; j += 32)
    out[(long long)row * D + j] = (_Float16)((xr[j] - mu) * rstd * g[j] + b[j]);
}

// ---------------------------------------------------------------------------
// GRU: persistent single workgroup (1024 thr = 32 waves), 49 sequential steps.
//   gi precomputed = xs@Wih^T + bih ([32*49][384] f32, sequence-major rows).
//   Each step: gh = h @ Whh^T via WMMA (M=32,N=384,K=128), gates in f32.
//   h kept in LDS as f16 (A-fragment friendly, ld=136 for 16B alignment).
//   Whh row-major [384][128] == the [N][K] layout -> B frags read from global.
// ---------------------------------------------------------------------------
__global__ void __launch_bounds__(1024) k_gru(
    const float* __restrict__ gi, const _Float16* __restrict__ whh,
    const float* __restrict__ bhh, float* __restrict__ ys)
{
  __shared__ float    gh[32 * 384];   // 48 KB
  __shared__ _Float16 hh[32 * 136];   // 8.5 KB
  const int tid = threadIdx.x;
  const int lane = tid & 31, wave = tid >> 5;
  const int l = lane & 15, hi = lane >> 4;

  for (int i = tid; i < 32 * 136; i += 1024) hh[i] = (_Float16)0.f;
  __syncthreads();

  for (int s = 0; s < 49; ++s) {
    // gh = h @ Whh^T : tiles (2 x 24) of 16x16, K loop 4 x 32
    for (int t = wave; t < 48; t += 32) {
      int mt = t & 1, nt = t >> 1;
      v8f accv = {};
#pragma unroll
      for (int k0 = 0; k0 < 128; k0 += 32) {
        FragH fa, fb;
        const _Float16* pa = hh + (mt * 16 + l) * 136 + k0 + hi * 8;
        fa.f4[0] = *(const float4*)pa;
        fa.f4[1] = *(const float4*)(pa + 16);
        const _Float16* pb = whh + (long long)(nt * 16 + l) * 128 + k0 + hi * 8;
        fb.f4[0] = *(const float4*)pb;
        fb.f4[1] = *(const float4*)(pb + 16);
        accv = wmma_f16f32(fa.v, fb.v, accv);
      }
#pragma unroll
      for (int r = 0; r < 8; ++r) {
        int gm = mt * 16 + r + hi * 8;
        int gn = nt * 16 + l;
        gh[gm * 384 + gn] = accv[r];
      }
    }
    __syncthreads();
    // Gate math (elementwise, 32x128)
    for (int i = tid; i < 32 * 128; i += 1024) {
      int bb = i >> 7, j = i & 127;
      const float* grow = gi + ((long long)bb * 49 + s) * 384;
      float ghr = gh[bb * 384 + j]       + bhh[j];
      float ghz = gh[bb * 384 + 128 + j] + bhh[128 + j];
      float ghn = gh[bb * 384 + 256 + j] + bhh[256 + j];
      float rr = 1.f / (1.f + expf(-(grow[j]       + ghr)));
      float zz = 1.f / (1.f + expf(-(grow[128 + j] + ghz)));
      float nn = tanhf(grow[256 + j] + rr * ghn);
      float hprev = (float)hh[bb * 136 + j];
      float hn = (1.f - zz) * nn + zz * hprev;
      ys[((long long)bb * 49 + s) * 128 + j] = hn;
      hh[bb * 136 + j] = (_Float16)hn;
    }
    __syncthreads();
  }
}

// ---------------------------------------------------------------------------
// Host launcher
// ---------------------------------------------------------------------------
extern "C" void kernel_launch(void* const* d_in, const int* in_sizes, int n_in,
                              void* d_out, int out_size, void* d_ws, size_t ws_size,
                              hipStream_t stream) {
  (void)in_sizes; (void)n_in; (void)out_size; (void)ws_size;

  const float* x     = (const float*)d_in[0];
  const float* lin1w = (const float*)d_in[31];
  const float* lin1b = (const float*)d_in[32];
  const float* ln1g  = (const float*)d_in[33];
  const float* ln1b  = (const float*)d_in[34];
  const float* Wih   = (const float*)d_in[35];
  const float* Whh   = (const float*)d_in[36];
  const float* bih   = (const float*)d_in[37];
  const float* bhh   = (const float*)d_in[38];
  const float* ln2g  = (const float*)d_in[39];
  const float* ln2b  = (const float*)d_in[40];
  const float* lin2w = (const float*)d_in[41];
  const float* lin2b = (const float*)d_in[42];

  char* ws = (char*)d_ws;
  size_t off = 0;
  auto alloc = [&](size_t bytes) -> size_t {
    off = (off + 255) & ~(size_t)255;
    size_t o = off; off += bytes; return o;
  };

  const int chans[6] = {3, 32, 64, 128, 256, 512};
  const int sp[6]    = {224, 112, 56, 28, 14, 7};

  size_t oD  = alloc((size_t)224 * 224 * 2);
  size_t oDT = alloc((size_t)224 * 224 * 2);
  size_t oWf[5], oBf[5];
  for (int lyr = 0; lyr < 5; ++lyr) {
    oWf[lyr] = alloc((size_t)chans[lyr + 1] * chans[lyr] * 9 * 2);
    oBf[lyr] = alloc((size_t)chans[lyr + 1] * 4);
  }
  size_t oLin1w = alloc((size_t)512 * 512 * 2);
  size_t oWih   = alloc((size_t)384 * 512 * 2);
  size_t oWhh   = alloc((size_t)384 * 128 * 2);
  size_t oLin2w = alloc((size_t)128 * 512 * 2);
  size_t oPing  = alloc((size_t)26  * 1024 * 1024);  // max: act1 = 25.7 MB
  size_t oPong  = alloc((size_t)104 * 1024 * 1024);  // max: conv1 out = 103 MB

  _Float16* Df16  = (_Float16*)(ws + oD);
  _Float16* DTf16 = (_Float16*)(ws + oDT);
  _Float16* ping  = (_Float16*)(ws + oPing);
  _Float16* pong  = (_Float16*)(ws + oPong);

  // --- prep ---
  k_build_dct<<<(224 * 224 + 255) / 256, 256, 0, stream>>>(Df16, DTf16);

  long long nx = (long long)32 * 3 * 224 * 224;
  k_f32_to_f16<<<(int)((nx + 255) / 256), 256, 0, stream>>>(x, ping, nx);

  for (int lyr = 0; lyr < 5; ++lyr) {
    int cin = chans[lyr], cout = chans[lyr + 1];
    const float* wv  = (const float*)d_in[1 + lyr * 6 + 0];
    const float* bv  = (const float*)d_in[1 + lyr * 6 + 1];
    const float* gv  = (const float*)d_in[1 + lyr * 6 + 2];
    const float* bev = (const float*)d_in[1 + lyr * 6 + 3];
    const float* mv  = (const float*)d_in[1 + lyr * 6 + 4];
    const float* vv  = (const float*)d_in[1 + lyr * 6 + 5];
    int tot = cout * cin * 9;
    k_fold_conv<<<(tot + 255) / 256, 256, 0, stream>>>(
        wv, bv, gv, bev, mv, vv,
        (_Float16*)(ws + oWf[lyr]), (float*)(ws + oBf[lyr]), cout, cin);
  }

  k_f32_to_f16<<<(512 * 512 + 255) / 256, 256, 0, stream>>>(lin1w, (_Float16*)(ws + oLin1w), 512 * 512);
  k_f32_to_f16<<<(384 * 512 + 255) / 256, 256, 0, stream>>>(Wih,   (_Float16*)(ws + oWih),   384 * 512);
  k_f32_to_f16<<<(384 * 128 + 255) / 256, 256, 0, stream>>>(Whh,   (_Float16*)(ws + oWhh),   384 * 128);
  k_f32_to_f16<<<(128 * 512 + 255) / 256, 256, 0, stream>>>(lin2w, (_Float16*)(ws + oLin2w), 128 * 512);

  // --- DCT: tmp = x @ D^T ; act0 = D @ tmp   (batch 96 of 224x224) ---
  {
    dim3 grid((224 + GT_M - 1) / GT_M, (224 + GT_N - 1) / GT_N, 96);
    k_gemm_wmma<<<grid, 256, 0, stream>>>(ping, DTf16, (void*)pong, nullptr,
        224, 224, 224, 224, 224, 224,
        224LL * 224, 0LL, 224LL * 224, 0, 0);
    k_gemm_wmma<<<grid, 256, 0, stream>>>(Df16, pong, (void*)ping, nullptr,
        224, 224, 224, 224, 224, 224,
        0LL, 224LL * 224, 224LL * 224, 0, 0);
  }

  // --- conv stack (conv+BN+GELU then 2x2 avgpool), ping <-> pong ---
  for (int lyr = 0; lyr < 5; ++lyr) {
    int cin = chans[lyr], cout = chans[lyr + 1];
    int Hc = sp[lyr], Wc = sp[lyr];
    int HWc = Hc * Wc;
    dim3 grid((cout + CV_M - 1) / CV_M, (HWc + CV_N - 1) / CV_N, 32);
    k_conv_wmma<<<grid, 256, 0, stream>>>(ping, (_Float16*)(ws + oWf[lyr]),
        (const float*)(ws + oBf[lyr]), pong, cin, cout, Hc, Wc);
    long long tot = (long long)32 * cout * (Hc / 2) * (Wc / 2);
    k_pool2x2<<<(int)((tot + 255) / 256), 256, 0, stream>>>(pong, ping, Hc, Wc, tot);
  }

  // --- head ---
  k_to_seq<<<(32 * 49 * 512 + 255) / 256, 256, 0, stream>>>(ping, pong);

  { // lin1: [1568,512] = seq @ W1[512,512] + b1  -> f32 in ping
    dim3 grid((1568 + GT_M - 1) / GT_M, (512 + GT_N - 1) / GT_N, 1);
    k_gemm_wmma<<<grid, 256, 0, stream>>>(pong, (_Float16*)(ws + oLin1w),
        (void*)ping, lin1b, 1568, 512, 512, 512, 512, 512, 0, 0, 0, 2, 0);
  }
  k_layernorm<<<(1568 + 7) / 8, 256, 0, stream>>>((const float*)ping, ln1g, ln1b,
      pong, 1568, 512);

  { // gi = ln1 @ Wih^T + bih  (Wih stored [384,512] == [N][K] -> btrans)
    dim3 grid((1568 + GT_M - 1) / GT_M, (384 + GT_N - 1) / GT_N, 1);
    k_gemm_wmma<<<grid, 256, 0, stream>>>(pong, (_Float16*)(ws + oWih),
        (void*)ping, bih, 1568, 384, 512, 512, 512, 384, 0, 0, 0, 2, 1);
  }

  k_gru<<<1, 1024, 0, stream>>>((const float*)ping, (_Float16*)(ws + oWhh), bhh,
      (float*)pong);

  k_layernorm<<<(1568 + 7) / 8, 256, 0, stream>>>((const float*)pong, ln2g, ln2b,
      ping, 1568, 128);

  { // lin2: [1568,512] = ln2 @ W2[128,512] + b2 -> d_out f32
    dim3 grid((1568 + GT_M - 1) / GT_M, (512 + GT_N - 1) / GT_N, 1);
    k_gemm_wmma<<<grid, 256, 0, stream>>>(ping, (_Float16*)(ws + oLin2w),
        d_out, lin2b, 1568, 512, 128, 128, 512, 512, 0, 0, 0, 2, 0);
  }
}
